// NeuralTensorDiagLayer_20375324852317
// MI455X (gfx1250) — matrix operand order repaired and checked by reading, not compiled, not run on hardware
//
#include <hip/hip_runtime.h>
#include <math.h>

typedef __bf16 bf16;
typedef __attribute__((ext_vector_type(16))) __bf16 v16bf;
typedef __attribute__((ext_vector_type(8)))  __bf16 v8bf;
typedef __attribute__((ext_vector_type(8)))  float  v8f;

// Problem sizes (fixed by the reference).
#define BSZ   8192   // batch
#define DDIM  2048   // d
#define KDIM  2048   // k (== d)
#define TWOD  4096   // 2d
#define NSTEP 128    // TWOD / 32 K-steps

#define LDSS  72     // LDS row stride in bf16 (144 B = 36 banks -> conflict-free 16-row b128 reads)

// ---------------------------------------------------------------------------
// Kernel 1: BvT[n][r] = (bf16) V[r][n]   (V: TWOD x KDIM f32, BvT: KDIM x TWOD bf16)
// ---------------------------------------------------------------------------
__global__ __launch_bounds__(256) void transpose_pack_v(const float* __restrict__ V,
                                                        bf16* __restrict__ BvT) {
    __shared__ float tile[32][33];
    const int n0 = blockIdx.x * 32;   // 0..2047
    const int r0 = blockIdx.y * 32;   // 0..4095
    const int tx = threadIdx.x;       // 0..31
    const int ty = threadIdx.y;       // 0..7
#pragma unroll
    for (int j = 0; j < 4; ++j) {
        tile[ty + j * 8][tx] = V[(size_t)(r0 + ty + j * 8) * KDIM + (n0 + tx)];
    }
    __syncthreads();
#pragma unroll
    for (int j = 0; j < 4; ++j) {
        BvT[(size_t)(n0 + ty + j * 8) * TWOD + (r0 + tx)] = (bf16)tile[tx][ty + j * 8];
    }
}

// ---------------------------------------------------------------------------
// Kernel 2: diag[t] = (1/KDIM) * sum_r e1[t,r]*e2[t,r]*W[t%DDIM, r]
// One wave32 per row; lane-strided float4 loads + shfl reduction.
// ---------------------------------------------------------------------------
__global__ __launch_bounds__(256) void diag_dot(const float* __restrict__ e1,
                                                const float* __restrict__ e2,
                                                const float* __restrict__ W,
                                                float* __restrict__ diag) {
    const int lane = threadIdx.x & 31;
    const int wave = threadIdx.x >> 5;
    const int row  = blockIdx.x * 8 + wave;           // 0..8191
    const int wrow = row & (DDIM - 1);
    const float* p1 = e1 + (size_t)row  * DDIM;
    const float* p2 = e2 + (size_t)row  * DDIM;
    const float* pw = W  + (size_t)wrow * DDIM;
    float s = 0.0f;
    for (int r = lane * 4; r < DDIM; r += 32 * 4) {
        float4 a = *(const float4*)(p1 + r);
        float4 b = *(const float4*)(p2 + r);
        float4 w = *(const float4*)(pw + r);
        s += a.x * b.x * w.x + a.y * b.y * w.y + a.z * b.z * w.z + a.w * b.w * w.w;
    }
#pragma unroll
    for (int off = 16; off >= 1; off >>= 1) s += __shfl_down(s, off, 32);
    if (lane == 0) diag[row] = s * (1.0f / (float)KDIM);
}

// ---------------------------------------------------------------------------
// Kernel 3: ff = [e1|e2] @ V via bf16 WMMA, double-buffered + async B staging.
//   out[i,j] = tanh(ff[i,j] + diag[(i&3)*DDIM + j] + bias[j])
// Workgroup: 256 threads (8 waves), 128x128 macro-tile; wave grid 2(M) x 4(N),
// each wave = 4x2 tiles of v_wmma_f32_16x16x32_bf16.
// ---------------------------------------------------------------------------
__global__ __launch_bounds__(256) void ff_gemm_tanh(const float* __restrict__ e1,
                                                    const float* __restrict__ e2,
                                                    const bf16*  __restrict__ BvT,
                                                    const float* __restrict__ diag,
                                                    const float* __restrict__ bias,
                                                    float* __restrict__ out) {
    __shared__ bf16 lds_a[2][128 * LDSS];
    __shared__ bf16 lds_b[2][128 * LDSS];

    const int tid  = threadIdx.x;
    const int lane = tid & 31;
    const int wave = tid >> 5;
    const int m0 = blockIdx.y * 128;
    const int n0 = blockIdx.x * 128;
    const int wm = (wave >> 2) * 64;  // wave M offset within macro-tile
    const int wn = (wave & 3) * 32;   // wave N offset within macro-tile

    // Fragment addressing (per ISA 16-bit A / B VGPR layouts, wave32)
    const int arow = lane & 15;
    const int ak   = ((lane >> 4) & 1) * 8;    // A: K chunks {ak..ak+7, ak+16..ak+23}
    const int bn   = lane & 15;
    const int bk   = ((lane >> 4) & 1) * 8;    // B: K chunks {bk..bk+7, bk+16..bk+23}

    // Issue the A-tile global loads for K-step `step` into registers (no LDS yet).
    auto loadA = [&](int step, float4* f) {
        const float* src = (step < NSTEP / 2) ? e1 : e2;
        const int kc = (step * 32) & (DDIM - 1);
#pragma unroll
        for (int p = 0; p < 4; ++p) {
            const int c   = tid + p * 256;     // 0..1023
            const int row = c >> 3;            // 0..127
            const int col = (c & 7) << 2;      // 0,4,...,28
            f[p] = *(const float4*)(src + (size_t)(m0 + row) * DDIM + kc + col);
        }
    };
    // Convert staged A registers to bf16 and store into LDS buffer `buf`.
    auto storeA = [&](int buf, const float4* f) {
#pragma unroll
        for (int p = 0; p < 4; ++p) {
            const int c   = tid + p * 256;
            const int row = c >> 3;
            const int col = (c & 7) << 2;
            union { bf16 h[4]; uint2 u; } cvt;
            cvt.h[0] = (bf16)f[p].x; cvt.h[1] = (bf16)f[p].y;
            cvt.h[2] = (bf16)f[p].z; cvt.h[3] = (bf16)f[p].w;
            *(uint2*)&lds_a[buf][row * LDSS + col] = cvt.u;
        }
    };
    // Async DMA the bf16 B tile straight into LDS (tracked by ASYNCcnt).
    auto asyncB = [&](int buf, int step) {
        const int kk = step * 32;
#pragma unroll
        for (int p = 0; p < 2; ++p) {
            const int c   = tid + p * 256;     // 0..511
            const int row = c >> 2;            // 0..127
            const int col = (c & 3) << 3;      // 0,8,16,24
            unsigned lofs = (unsigned)(size_t)&lds_b[buf][row * LDSS + col];
            unsigned long long gptr =
                (unsigned long long)(BvT + (size_t)(n0 + row) * TWOD + kk + col);
            asm volatile("global_load_async_to_lds_b128 %0, %1, off"
                         :: "v"(lofs), "v"(gptr) : "memory");
        }
    };

    v8f acc[4][2];
#pragma unroll
    for (int mi = 0; mi < 4; ++mi)
#pragma unroll
        for (int ni = 0; ni < 2; ++ni) acc[mi][ni] = (v8f){0.f,0.f,0.f,0.f,0.f,0.f,0.f,0.f};

    // ---- prologue: stage step 0 ----
    float4 fa[4];
    loadA(0, fa);
    asyncB(0, 0);
    storeA(0, fa);
    asm volatile("s_wait_asynccnt 0" ::: "memory");
    __syncthreads();

    for (int step = 0; step < NSTEP; ++step) {
        const int cur = step & 1;
        const int nxt = cur ^ 1;
        if (step + 1 < NSTEP) {
            loadA(step + 1, fa);     // global loads in flight during WMMAs
            asyncB(nxt, step + 1);   // async DMA into the other buffer
        }

        // ---- fragments + WMMA from current buffer ----
        v16bf af[4], bfrag[2];
#pragma unroll
        for (int mi = 0; mi < 4; ++mi) {
            const bf16* pa = &lds_a[cur][(wm + mi * 16 + arow) * LDSS + ak];
            v8bf lo = *(const v8bf*)(pa);
            v8bf hi = *(const v8bf*)(pa + 16);
            af[mi] = __builtin_shufflevector(lo, hi, 0,1,2,3,4,5,6,7,8,9,10,11,12,13,14,15);
        }
#pragma unroll
        for (int ni = 0; ni < 2; ++ni) {
            const bf16* pb = &lds_b[cur][(wn + ni * 16 + bn) * LDSS + bk];
            v8bf lo = *(const v8bf*)(pb);
            v8bf hi = *(const v8bf*)(pb + 16);
            bfrag[ni] = __builtin_shufflevector(lo, hi, 0,1,2,3,4,5,6,7,8,9,10,11,12,13,14,15);
        }
#pragma unroll
        for (int mi = 0; mi < 4; ++mi)
#pragma unroll
            for (int ni = 0; ni < 2; ++ni)
                acc[mi][ni] = __builtin_amdgcn_wmma_f32_16x16x32_bf16(
                    false, af[mi], false, bfrag[ni], (short)0, acc[mi][ni], false, false);

        if (step + 1 < NSTEP) {
            storeA(nxt, fa);         // convert+store after WMMAs issued
        }
        asm volatile("s_wait_asynccnt 0" ::: "memory");
        __syncthreads();
    }

    // ---- epilogue: out = tanh(acc + diag + bias), per C-matrix lane layout ----
#pragma unroll
    for (int mi = 0; mi < 4; ++mi) {
#pragma unroll
        for (int ni = 0; ni < 2; ++ni) {
            const int gi0 = m0 + wm + mi * 16 + ((lane >> 4) & 1) * 8;
            const int gj  = n0 + wn + ni * 16 + (lane & 15);
            const float bj = bias[gj];
#pragma unroll
            for (int v = 0; v < 8; ++v) {
                const int gi = gi0 + v;
                float val = acc[mi][ni][v] + diag[((gi & 3) << 11) + gj] + bj;
                out[(size_t)gi * KDIM + gj] = tanhf(val);
            }
        }
    }
}

// ---------------------------------------------------------------------------
extern "C" void kernel_launch(void* const* d_in, const int* in_sizes, int n_in,
                              void* d_out, int out_size, void* d_ws, size_t ws_size,
                              hipStream_t stream) {
    const float* e1 = (const float*)d_in[0];
    const float* e2 = (const float*)d_in[1];
    const float* W  = (const float*)d_in[2];
    const float* V  = (const float*)d_in[3];
    const float* b  = (const float*)d_in[4];
    float* out = (float*)d_out;

    // workspace layout: BvT (KDIM x TWOD bf16 = 16 MB) | diag (8192 f32)
    bf16*  BvT  = (bf16*)d_ws;
    float* diag = (float*)((char*)d_ws + (size_t)KDIM * TWOD * sizeof(bf16));

    transpose_pack_v<<<dim3(KDIM / 32, TWOD / 32), dim3(32, 8), 0, stream>>>(V, BvT);
    diag_dot<<<BSZ / 8, 256, 0, stream>>>(e1, e2, W, diag);
    ff_gemm_tanh<<<dim3(KDIM / 128, BSZ / 128), 256, 0, stream>>>(e1, e2, BvT, diag, b, out);
}
